// Qwen3_6SparseMoeBlock_89764816486465
// MI455X (gfx1250) — matrix operand order repaired and checked by reading, not compiled, not run on hardware
//
#include <hip/hip_runtime.h>
#include <hip/hip_bf16.h>

// ---------------------------------------------------------------------------
// Qwen3 sparse MoE forward for gfx1250 (MI455X), bf16 WMMA path.
//   HIDDEN=2048, NUM_EXPERTS=32, TOP_K=4, D_MOE=768, D_SHARED=4096, T=2048
// GEMMs: A = bf16 activations, double-buffered in LDS via pipelined
// global_load_async_to_lds_b64 (uniform EXEC, 1 chunk/thread); B = fp32
// weights (row-major NxK) converted to bf16 in-register (v_perm_b32 packing);
// f32 accumulation via v_wmma_f32_16x16x32_bf16.  Each wave register-blocks
// 4 M-tiles x 1 N-tile to amortize B conversion.
// ---------------------------------------------------------------------------

#define HIDDEN     2048
#define NUM_EXPERTS  32
#define TOP_K         4
#define D_MOE       768
#define D_SHARED   4096
#define T_TOKENS   2048
#define MAXTOK     T_TOKENS

// LDS A tile: 64 rows x 32 k bf16, row stride padded to 40 ushorts (80B:
// 16B-aligned for b128 fragment reads, bank-conflict-free across 16 rows).
#define A_STRIDE 40
#define A_TILE_USHORT (64 * A_STRIDE)          // one buffer, ushort units
#define A_TILE_BYTES  (A_TILE_USHORT * 2)      // 5120 B

typedef __attribute__((ext_vector_type(16))) __bf16       v16bf;
typedef __attribute__((ext_vector_type(8)))  float        v8f;
typedef __attribute__((ext_vector_type(4)))  unsigned int u32x4;
typedef __attribute__((ext_vector_type(4)))  float        f32x4;

__device__ __forceinline__ unsigned short f2bf(float f) {
  return (unsigned short)((__float_as_uint(f) + 0x8000u) >> 16);  // round-nearest
}
// Pack two fp32 -> packed bf16 pair: two v_add + one v_perm_b32.
__device__ __forceinline__ unsigned int pack2bf(float lo, float hi) {
  unsigned int ulo = __float_as_uint(lo) + 0x8000u;
  unsigned int uhi = __float_as_uint(hi) + 0x8000u;
  return __builtin_amdgcn_perm(uhi, ulo, 0x07060302u);
}
__device__ __forceinline__ float fast_sigmoid(float x) {
  return __builtin_amdgcn_rcpf(1.f + __expf(-x));       // v_rcp_f32, no IEEE div
}
__device__ __forceinline__ float fast_silu(float x) { return x * fast_sigmoid(x); }

// --- CDNA5 async copy: 8B global -> LDS, tracked by ASYNCcnt. ---
__device__ __forceinline__ void async_ld_b64(unsigned lds_off, const void* g) {
  asm volatile("global_load_async_to_lds_b64 %0, %1, off"
               :: "v"(lds_off), "v"(g) : "memory");
}
__device__ __forceinline__ void wait_async0() {
  asm volatile("s_wait_asynccnt 0" ::: "memory");
}

// --- A fragment (16x32 bf16) from LDS tile.  this code's k order: lane row = lane%16,
//     16 contiguous elems at K = g*16 (g = lane/16).
__device__ __forceinline__ v16bf lds_a_frag(const unsigned short* __restrict__ tile,
                                            int row, int lane) {
  int g = (lane >> 4) & 1;
  const unsigned short* p = tile + row * A_STRIDE + g * 16;
  union { v16bf v; u32x4 q[2]; } fr;
  fr.q[0] = *(const u32x4*)(p);        // ds_load_b128
  fr.q[1] = *(const u32x4*)(p + 8);    // ds_load_b128
  return fr.v;
}

// --- B fragment (32x16 bf16): lane = column N; 16 contiguous fp32 K-values of
//     weight row n, converted to bf16 pairs in-register; prefetch next K chunk.
__device__ __forceinline__ v16bf load_b_f32(const float* __restrict__ W,
                                            int n0, int ldk, int k0, int lane) {
  int col = lane & 15, g = (lane >> 4) & 1;
  const float* p = W + (long long)(n0 + col) * ldk + k0 + g * 16;
  f32x4 f0 = *(const f32x4*)(p);
  f32x4 f1 = *(const f32x4*)(p + 4);
  f32x4 f2 = *(const f32x4*)(p + 8);
  f32x4 f3 = *(const f32x4*)(p + 12);
  __builtin_prefetch((const char*)p + 128, 0, 0);   // global_prefetch_b8, next k0
  union { v16bf v; unsigned int u[8]; } fr;
  fr.u[0] = pack2bf(f0[0], f0[1]);
  fr.u[1] = pack2bf(f0[2], f0[3]);
  fr.u[2] = pack2bf(f1[0], f1[1]);
  fr.u[3] = pack2bf(f1[2], f1[3]);
  fr.u[4] = pack2bf(f2[0], f2[1]);
  fr.u[5] = pack2bf(f2[2], f2[3]);
  fr.u[6] = pack2bf(f3[0], f3[1]);
  fr.u[7] = pack2bf(f3[2], f3[3]);
  return fr.v;
}

__device__ __forceinline__ v8f wmma_bf16(v16bf a, v16bf b, v8f c) {
  return __builtin_amdgcn_wmma_f32_16x16x32_bf16(false, a, false, b, (short)0, c,
                                                 false, false);
}

// ---------------------------------------------------------------------------
__global__ void init_counts_kernel(int* __restrict__ counts) {
  counts[threadIdx.x] = 0;
}

// One block per token: router logits, softmax, top-4 -> per-expert lists;
// shared-expert sigmoid gate; x row -> bf16.
__global__ void router_kernel(const float* __restrict__ x,
                              const float* __restrict__ gate_w,
                              const float* __restrict__ seg_w,
                              unsigned short* __restrict__ x_bf,
                              float* __restrict__ sgate,
                              int* __restrict__ counts,
                              int* __restrict__ list_entry,
                              float* __restrict__ list_w) {
  int t = blockIdx.x;
  int tid = threadIdx.x;
  const float* xr = x + (long long)t * HIDDEN;

  __shared__ float part[NUM_EXPERTS][8];
  __shared__ float red[256];
  __shared__ float logits[NUM_EXPERTS];

  int e = tid & 31, kp = tid >> 5;
  const float* w = gate_w + (long long)e * HIDDEN;
  float acc = 0.f;
  for (int k = kp * 256; k < kp * 256 + 256; ++k) acc += xr[k] * w[k];
  part[e][kp] = acc;

  float acc2 = 0.f;
#pragma unroll
  for (int k = 0; k < 8; ++k) {
    float v = xr[tid * 8 + k];
    acc2 += v * seg_w[tid * 8 + k];
    x_bf[(long long)t * HIDDEN + tid * 8 + k] = f2bf(v);
  }
  red[tid] = acc2;
  __syncthreads();
  for (int s = 128; s > 0; s >>= 1) {
    if (tid < s) red[tid] += red[tid + s];
    __syncthreads();
  }
  if (tid < NUM_EXPERTS) {
    float s = 0.f;
#pragma unroll
    for (int i = 0; i < 8; ++i) s += part[tid][i];
    logits[tid] = s;
  }
  __syncthreads();

  if (tid == 0) {
    sgate[t] = fast_sigmoid(red[0]);
    float mx = -3.4e38f;
    for (int i = 0; i < NUM_EXPERTS; ++i) mx = fmaxf(mx, logits[i]);
    float pr[NUM_EXPERTS];
    float sum = 0.f;
    for (int i = 0; i < NUM_EXPERTS; ++i) { pr[i] = __expf(logits[i] - mx); sum += pr[i]; }
    float inv = __builtin_amdgcn_rcpf(sum);
    for (int i = 0; i < NUM_EXPERTS; ++i) pr[i] *= inv;
    int   idx[TOP_K];
    float wv[TOP_K];
    float wsum = 0.f;
    for (int j = 0; j < TOP_K; ++j) {
      int bi = 0; float bv = -1.f;
      for (int i = 0; i < NUM_EXPERTS; ++i)
        if (pr[i] > bv) { bv = pr[i]; bi = i; }
      idx[j] = bi; wv[j] = bv; wsum += bv; pr[bi] = -1.f;
    }
    float invw = 1.f / (wsum + 1e-20f);
    for (int j = 0; j < TOP_K; ++j) {
      int slot = atomicAdd(&counts[idx[j]], 1);
      list_entry[idx[j] * MAXTOK + slot] = (t << 2) | j;   // h row = t*4+j
      list_w[idx[j] * MAXTOK + slot]     = wv[j] * invw;
    }
  }
}

// ---------------------------------------------------------------------------
// Pipelined GEMM body shared by all four WMMA kernels.
//   Block: 512 thr = 16 waves; tile M=64 (4 m-subtiles/wave) x N=256.
//   A tile double-buffered in LDS; tile i+1 copied asynchronously (b64/thread)
//   while tile i is consumed; one barrier per K step.
// Staging slot per thread: chunk c = tid in [0,512): row = c/8, 8B seg = c%8.

#define GEMM_PIPELINE(K_TOTAL, LOAD_B_AND_WMMA)                                \
  {                                                                            \
    const int nk = (K_TOTAL) / 32;                                             \
    async_ld_b64(my_lds, my_g);                    /* prologue: tile 0 */      \
    wait_async0();                                                             \
    __syncthreads();                                                           \
    for (int i = 0; i < nk; ++i) {                                             \
      if (i + 1 < nk)                              /* pipeline: tile i+1 */    \
        async_ld_b64(my_lds + ((unsigned)(i + 1) & 1u) * A_TILE_BYTES,         \
                     my_g + (i + 1) * 32);                                     \
      const unsigned short* curA = lbA + (i & 1) * A_TILE_USHORT;              \
      int k0 = i * 32;                                                         \
      v16bf a0 = lds_a_frag(curA,      (lane & 15), lane);                     \
      v16bf a1 = lds_a_frag(curA, 16 + (lane & 15), lane);                     \
      v16bf a2 = lds_a_frag(curA, 32 + (lane & 15), lane);                     \
      v16bf a3 = lds_a_frag(curA, 48 + (lane & 15), lane);                     \
      LOAD_B_AND_WMMA                                                          \
      wait_async0();                                                           \
      __syncthreads();                                                         \
    }                                                                          \
  }

// sh = silu(x @ Wg^T) * (x @ Wu^T) -> bf16 (T x D_SHARED)
__global__ void shared_mlp1_kernel(const unsigned short* __restrict__ x_bf,
                                   const float* __restrict__ Wg,
                                   const float* __restrict__ Wu,
                                   unsigned short* __restrict__ sh_bf) {
  __shared__ unsigned short lbA[2 * A_TILE_USHORT];
  int tid = threadIdx.x;
  int lane = tid & 31, wave = tid >> 5;
  int m0 = blockIdx.x * 64;
  int n0 = (blockIdx.y * 16 + wave) * 16;

  unsigned lds_base = (unsigned)(unsigned long long)(void*)lbA;
  int srow = tid >> 3, sseg = tid & 7;           // 512 x 8B chunks
  unsigned my_lds = lds_base + (unsigned)(srow * A_STRIDE * 2 + sseg * 8);
  const unsigned short* my_g = x_bf + (long long)(m0 + srow) * HIDDEN + sseg * 4;

  v8f cg[4] = {}; v8f cu[4] = {};
  GEMM_PIPELINE(HIDDEN, {
    v16bf bg = load_b_f32(Wg, n0, HIDDEN, k0, lane);
    v16bf bu = load_b_f32(Wu, n0, HIDDEN, k0, lane);
    cg[0] = wmma_bf16(a0, bg, cg[0]);  cu[0] = wmma_bf16(a0, bu, cu[0]);
    cg[1] = wmma_bf16(a1, bg, cg[1]);  cu[1] = wmma_bf16(a1, bu, cu[1]);
    cg[2] = wmma_bf16(a2, bg, cg[2]);  cu[2] = wmma_bf16(a2, bu, cu[2]);
    cg[3] = wmma_bf16(a3, bg, cg[3]);  cu[3] = wmma_bf16(a3, bu, cu[3]);
  })

  int col = lane & 15, mrow = (lane >> 4) * 8;
#pragma unroll
  for (int s = 0; s < 4; ++s)
#pragma unroll
    for (int r = 0; r < 8; ++r) {
      float h = fast_silu(cg[s][r]) * cu[s][r];
      long long row = m0 + s * 16 + mrow + r;
      sh_bf[row * (long long)D_SHARED + n0 + col] = f2bf(h);
    }
}

// out = sigmoid_gate[t] * (sh @ Wd^T)   (writes every element of d_out)
__global__ void shared_mlp2_kernel(const unsigned short* __restrict__ sh_bf,
                                   const float* __restrict__ Wd,
                                   const float* __restrict__ sgate,
                                   float* __restrict__ out) {
  __shared__ unsigned short lbA[2 * A_TILE_USHORT];
  int tid = threadIdx.x;
  int lane = tid & 31, wave = tid >> 5;
  int m0 = blockIdx.x * 64;
  int n0 = (blockIdx.y * 16 + wave) * 16;

  unsigned lds_base = (unsigned)(unsigned long long)(void*)lbA;
  int srow = tid >> 3, sseg = tid & 7;
  unsigned my_lds = lds_base + (unsigned)(srow * A_STRIDE * 2 + sseg * 8);
  const unsigned short* my_g = sh_bf + (long long)(m0 + srow) * D_SHARED + sseg * 4;

  v8f c[4] = {};
  GEMM_PIPELINE(D_SHARED, {
    v16bf b = load_b_f32(Wd, n0, D_SHARED, k0, lane);
    c[0] = wmma_bf16(a0, b, c[0]);
    c[1] = wmma_bf16(a1, b, c[1]);
    c[2] = wmma_bf16(a2, b, c[2]);
    c[3] = wmma_bf16(a3, b, c[3]);
  })

  int col = lane & 15, mrow = (lane >> 4) * 8;
#pragma unroll
  for (int s = 0; s < 4; ++s)
#pragma unroll
    for (int r = 0; r < 8; ++r) {
      long long row = m0 + s * 16 + mrow + r;
      out[row * (long long)HIDDEN + n0 + col] = sgate[row] * c[s][r];
    }
}

// Per-expert gate_up with gathered token rows; h = silu(g)*u -> bf16 h_buf.
// grid (D_MOE/256, T/64, NUM_EXPERTS); 64-slot tiles, clamp loads/mask stores.
__global__ void moe_gateup_kernel(const unsigned short* __restrict__ x_bf,
                                  const float* __restrict__ gate_up_proj,
                                  const int* __restrict__ counts,
                                  const int* __restrict__ list_entry,
                                  unsigned short* __restrict__ h_buf) {
  __shared__ unsigned short lbA[2 * A_TILE_USHORT];
  int e = blockIdx.z;
  int cnt = counts[e];
  int slot0 = blockIdx.y * 64;
  if (slot0 >= cnt) return;                       // block-uniform, before barriers
  int tid = threadIdx.x;
  int lane = tid & 31, wave = tid >> 5;
  int n0 = (blockIdx.x * 16 + wave) * 16;         // [0, D_MOE)
  const int* lst = list_entry + e * MAXTOK;

  unsigned lds_base = (unsigned)(unsigned long long)(void*)lbA;
  int srow = tid >> 3, sseg = tid & 7;
  unsigned my_lds = lds_base + (unsigned)(srow * A_STRIDE * 2 + sseg * 8);
  long long trow = (long long)(lst[min(slot0 + srow, cnt - 1)] >> 2);
  const unsigned short* my_g = x_bf + trow * HIDDEN + sseg * 4;

  const float* Wgp = gate_up_proj + (long long)e * (2 * D_MOE) * HIDDEN;
  const float* Wup = Wgp + (long long)D_MOE * HIDDEN;

  v8f cg[4] = {}; v8f cu[4] = {};
  GEMM_PIPELINE(HIDDEN, {
    v16bf bg = load_b_f32(Wgp, n0, HIDDEN, k0, lane);
    v16bf bu = load_b_f32(Wup, n0, HIDDEN, k0, lane);
    cg[0] = wmma_bf16(a0, bg, cg[0]);  cu[0] = wmma_bf16(a0, bu, cu[0]);
    cg[1] = wmma_bf16(a1, bg, cg[1]);  cu[1] = wmma_bf16(a1, bu, cu[1]);
    cg[2] = wmma_bf16(a2, bg, cg[2]);  cu[2] = wmma_bf16(a2, bu, cu[2]);
    cg[3] = wmma_bf16(a3, bg, cg[3]);  cu[3] = wmma_bf16(a3, bu, cu[3]);
  })

  int col = lane & 15, mrow = (lane >> 4) * 8;
#pragma unroll
  for (int s = 0; s < 4; ++s)
#pragma unroll
    for (int r = 0; r < 8; ++r) {
      int slot = slot0 + s * 16 + mrow + r;
      if (slot < cnt) {
        int ent = lst[slot];
        float h = fast_silu(cg[s][r]) * cu[s][r];
        h_buf[(long long)ent * D_MOE + n0 + col] = f2bf(h);
      }
    }
}

// Per-expert down-proj; y scaled by routing weight, atomically added to out.
// grid (HIDDEN/256, T/64, NUM_EXPERTS)
__global__ void moe_down_kernel(const unsigned short* __restrict__ h_buf,
                                const float* __restrict__ down_proj,
                                const int* __restrict__ counts,
                                const int* __restrict__ list_entry,
                                const float* __restrict__ list_w,
                                float* __restrict__ out) {
  __shared__ unsigned short lbA[2 * A_TILE_USHORT];
  int e = blockIdx.z;
  int cnt = counts[e];
  int slot0 = blockIdx.y * 64;
  if (slot0 >= cnt) return;
  int tid = threadIdx.x;
  int lane = tid & 31, wave = tid >> 5;
  int n0 = (blockIdx.x * 16 + wave) * 16;         // [0, HIDDEN)
  const int* lst = list_entry + e * MAXTOK;
  const float* lw = list_w + e * MAXTOK;

  unsigned lds_base = (unsigned)(unsigned long long)(void*)lbA;
  int srow = tid >> 3, sseg = tid & 7;
  unsigned my_lds = lds_base + (unsigned)(srow * A_STRIDE * 2 + sseg * 8);
  long long hrow = (long long)lst[min(slot0 + srow, cnt - 1)];
  const unsigned short* my_g = h_buf + hrow * D_MOE + sseg * 4;

  const float* Wd = down_proj + (long long)e * HIDDEN * D_MOE;

  v8f c[4] = {};
  GEMM_PIPELINE(D_MOE, {
    v16bf b = load_b_f32(Wd, n0, D_MOE, k0, lane);
    c[0] = wmma_bf16(a0, b, c[0]);
    c[1] = wmma_bf16(a1, b, c[1]);
    c[2] = wmma_bf16(a2, b, c[2]);
    c[3] = wmma_bf16(a3, b, c[3]);
  })

  int col = lane & 15, mrow = (lane >> 4) * 8;
#pragma unroll
  for (int s = 0; s < 4; ++s)
#pragma unroll
    for (int r = 0; r < 8; ++r) {
      int slot = slot0 + s * 16 + mrow + r;
      if (slot < cnt) {
        int ent = lst[slot];
        int t = ent >> 2;
        atomicAdd(&out[(long long)t * HIDDEN + n0 + col], lw[slot] * c[s][r]);
      }
    }
}

// ---------------------------------------------------------------------------
extern "C" void kernel_launch(void* const* d_in, const int* in_sizes, int n_in,
                              void* d_out, int out_size, void* d_ws, size_t ws_size,
                              hipStream_t stream) {
  (void)in_sizes; (void)n_in; (void)out_size; (void)ws_size;
  const float* x             = (const float*)d_in[0];
  const float* gate_w        = (const float*)d_in[1];
  const float* gate_up_proj  = (const float*)d_in[2];
  const float* down_proj     = (const float*)d_in[3];
  const float* shared_gate_w = (const float*)d_in[4];
  const float* shared_up_w   = (const float*)d_in[5];
  const float* shared_down_w = (const float*)d_in[6];
  const float* shared_eg_w   = (const float*)d_in[7];
  float* out = (float*)d_out;

  char* ws = (char*)d_ws;
  size_t off = 0;
  auto take = [&](size_t bytes) -> char* {
    char* p = ws + off;
    off = (off + bytes + 255) & ~(size_t)255;
    return p;
  };
  unsigned short* x_bf   = (unsigned short*)take((size_t)T_TOKENS * HIDDEN * 2);          //  8 MB
  unsigned short* sh_bf  = (unsigned short*)take((size_t)T_TOKENS * D_SHARED * 2);        // 16 MB
  unsigned short* h_buf  = (unsigned short*)take((size_t)T_TOKENS * TOP_K * D_MOE * 2);   // 12 MB
  float*          sgate  = (float*)take((size_t)T_TOKENS * 4);
  int*            counts = (int*)take((size_t)NUM_EXPERTS * 4);
  int*            l_ent  = (int*)take((size_t)NUM_EXPERTS * MAXTOK * 4);
  float*          l_w    = (float*)take((size_t)NUM_EXPERTS * MAXTOK * 4);

  init_counts_kernel<<<1, NUM_EXPERTS, 0, stream>>>(counts);
  router_kernel<<<T_TOKENS, 256, 0, stream>>>(x, gate_w, shared_eg_w, x_bf, sgate,
                                              counts, l_ent, l_w);
  shared_mlp1_kernel<<<dim3(T_TOKENS / 64, D_SHARED / 256), 512, 0, stream>>>(
      x_bf, shared_gate_w, shared_up_w, sh_bf);
  shared_mlp2_kernel<<<dim3(T_TOKENS / 64, HIDDEN / 256), 512, 0, stream>>>(
      sh_bf, shared_down_w, sgate, out);
  moe_gateup_kernel<<<dim3(D_MOE / 256, T_TOKENS / 64, NUM_EXPERTS), 512, 0, stream>>>(
      x_bf, gate_up_proj, counts, l_ent, h_buf);
  moe_down_kernel<<<dim3(HIDDEN / 256, T_TOKENS / 64, NUM_EXPERTS), 512, 0, stream>>>(
      h_buf, down_proj, counts, l_ent, l_w, out);
}
